// Aggregator_63015760167156
// MI455X (gfx1250) — matrix operand order repaired and checked by reading, not compiled, not run on hardware
//
#include <hip/hip_runtime.h>

#define N_NODES 100000
#define N_EDGES 3200000
#define D       256

typedef __attribute__((ext_vector_type(16))) __bf16 v16bf;
typedef __attribute__((ext_vector_type(8)))  float  v8f;

// ---------- helpers ----------
static __device__ __forceinline__ unsigned int f2bf_pk(float lo, float hi) {
    // round-to-nearest-even f32 -> bf16, packed (lo in [15:0], hi in [31:16])
    unsigned int ul = __float_as_uint(lo);
    unsigned int uh = __float_as_uint(hi);
    ul = (ul + 0x7fffu + ((ul >> 16) & 1u)) >> 16;
    uh = (uh + 0x7fffu + ((uh >> 16) & 1u)) >> 16;
    return (ul & 0xffffu) | (uh << 16);
}

static __device__ __forceinline__ v16bf make_frag(uint4 lo, uint4 hi) {
    union { uint4 q[2]; v16bf v; } u;
    u.q[0] = lo; u.q[1] = hi;
    return u.v;
}

// ---------- kernel 0: zero the side accumulator (d_out reused as scratch) ----------
__global__ void zero_f4(float4* __restrict__ p, int n4) {
    int i = blockIdx.x * blockDim.x + threadIdx.x;
    if (i < n4) p[i] = make_float4(0.f, 0.f, 0.f, 0.f);
}

// ---------- kernel 1: convert W1,W2 (row-major [256][256] f32) to bf16 ----------
__global__ void wconv_bf16(const float* __restrict__ W1, const float* __restrict__ W2,
                           unsigned short* __restrict__ wb) {
    int t = blockIdx.x * blockDim.x + threadIdx.x;   // 0..16383
    int base = t * 8;                                 // element index in combined 2*64K space
    const float* src;
    int off;
    if (base < D * D) { src = W1; off = base; }
    else              { src = W2; off = base - D * D; }
    float4 a = *reinterpret_cast<const float4*>(src + off);
    float4 b = *reinterpret_cast<const float4*>(src + off + 4);
    uint4 o;
    o.x = f2bf_pk(a.x, a.y);
    o.y = f2bf_pk(a.z, a.w);
    o.z = f2bf_pk(b.x, b.y);
    o.w = f2bf_pk(b.z, b.w);
    *reinterpret_cast<uint4*>(wb + base) = o;
}

// ---------- kernel 2: SpMM  side[r] += val * ego[c]  (one wave per edge) ----------
__global__ void spmm_atomic(const float* __restrict__ ego,
                            const float* __restrict__ vals,
                            const int*   __restrict__ rows,
                            const int*   __restrict__ cols,
                            float*       __restrict__ side) {
    unsigned int lin  = blockIdx.x * blockDim.x + threadIdx.x;
    unsigned int wid  = lin >> 5;          // edge id
    unsigned int lane = lin & 31u;
    if (wid >= N_EDGES) return;
    int   r = rows[wid];
    int   c = cols[wid];
    float v = vals[wid];
    const float* srow = ego  + (size_t)c * D + lane * 8;
    float*       drow = side + (size_t)r * D + lane * 8;
    float4 a = *reinterpret_cast<const float4*>(srow);
    float4 b = *reinterpret_cast<const float4*>(srow + 4);
    atomicAdd(drow + 0, a.x * v);
    atomicAdd(drow + 1, a.y * v);
    atomicAdd(drow + 2, a.z * v);
    atomicAdd(drow + 3, a.w * v);
    atomicAdd(drow + 4, b.x * v);
    atomicAdd(drow + 5, b.y * v);
    atomicAdd(drow + 6, b.z * v);
    atomicAdd(drow + 7, b.w * v);
}

// ---------- kernel 3: fused bi-interaction, bf16 WMMA ----------
// io holds `side` on entry; overwritten with the final output (per-tile, after barrier).
__global__ void __launch_bounds__(512)
fused_bi_wmma(const float* __restrict__ ego,
              float*       __restrict__ io,
              const unsigned short* __restrict__ wb,   // [0..64K): W1 bf16, [64K..128K): W2 bf16
              const float* __restrict__ b1,
              const float* __restrict__ b2) {
    __shared__ alignas(16) unsigned short As[16 * D];  // bf16(ego + side) tile
    __shared__ alignas(16) unsigned short Ab[16 * D];  // bf16(ego * side) tile

    const int tid  = threadIdx.x;
    const int row0 = blockIdx.x * 16;

    // ---- phase 1: stage 16x256 sum/prod tiles into LDS as bf16 ----
    {
        int r = tid >> 5;             // 0..15
        int c = (tid & 31) * 8;       // 0..248
        const float* e = ego + (size_t)(row0 + r) * D + c;
        const float* s = io  + (size_t)(row0 + r) * D + c;
        float4 e0 = *reinterpret_cast<const float4*>(e);
        float4 e1 = *reinterpret_cast<const float4*>(e + 4);
        float4 s0 = *reinterpret_cast<const float4*>(s);
        float4 s1 = *reinterpret_cast<const float4*>(s + 4);
        uint4 us, up;
        us.x = f2bf_pk(e0.x + s0.x, e0.y + s0.y);
        us.y = f2bf_pk(e0.z + s0.z, e0.w + s0.w);
        us.z = f2bf_pk(e1.x + s1.x, e1.y + s1.y);
        us.w = f2bf_pk(e1.z + s1.z, e1.w + s1.w);
        up.x = f2bf_pk(e0.x * s0.x, e0.y * s0.y);
        up.y = f2bf_pk(e0.z * s0.z, e0.w * s0.w);
        up.z = f2bf_pk(e1.x * s1.x, e1.y * s1.y);
        up.w = f2bf_pk(e1.z * s1.z, e1.w * s1.w);
        *reinterpret_cast<uint4*>(As + r * D + c) = us;
        *reinterpret_cast<uint4*>(Ab + r * D + c) = up;
    }
    __syncthreads();

    // ---- phase 2: each wave owns one 16-column block; 8 K-steps of dual WMMA ----
    const int wave = tid >> 5;        // n-block 0..15
    const int lane = tid & 31;
    const int half = lane >> 4;       // 0/1
    const int n    = lane & 15;
    const int col  = wave * 16 + n;
    const int m    = n;               // A-matrix row handled by this lane

    const unsigned short* w1b = wb;
    const unsigned short* w2b = wb + D * D;

    v8f acc1 = {}; v8f acc2 = {};

    #pragma unroll
    for (int kb = 0; kb < 8; ++kb) {
        const int k1 = kb * 32 + half * 16;       // A elems 0..7
        const int k2 = k1 + 8;                    // A elems 8..15
        uint4 a0 = *reinterpret_cast<const uint4*>(As + m * D + k1);
        uint4 a1 = *reinterpret_cast<const uint4*>(As + m * D + k2);
        uint4 p0 = *reinterpret_cast<const uint4*>(Ab + m * D + k1);
        uint4 p1 = *reinterpret_cast<const uint4*>(Ab + m * D + k2);
        v16bf a_sum = make_frag(a0, a1);
        v16bf a_bi  = make_frag(p0, p1);

        const int kw = kb * 32 + half * 16;       // B elems 0..15 (contiguous K)
        const unsigned short* q1 = w1b + col * D + kw;
        const unsigned short* q2 = w2b + col * D + kw;
        uint4 w1lo = *reinterpret_cast<const uint4*>(q1);
        uint4 w1hi = *reinterpret_cast<const uint4*>(q1 + 8);
        uint4 w2lo = *reinterpret_cast<const uint4*>(q2);
        uint4 w2hi = *reinterpret_cast<const uint4*>(q2 + 8);
        v16bf bw1 = make_frag(w1lo, w1hi);
        v16bf bw2 = make_frag(w2lo, w2hi);

        acc1 = __builtin_amdgcn_wmma_f32_16x16x32_bf16(
                   false, a_sum, false, bw1, (short)0, acc1, false, false);
        acc2 = __builtin_amdgcn_wmma_f32_16x16x32_bf16(
                   false, a_bi,  false, bw2, (short)0, acc2, false, false);
    }

    // ---- phase 3: bias + leaky_relu(0.01) + sum, write f32 ----
    const float bias1 = b1[col];
    const float bias2 = b2[col];
    #pragma unroll
    for (int r = 0; r < 8; ++r) {
        float s = acc1[r] + bias1; s = (s > 0.f) ? s : 0.01f * s;
        float t = acc2[r] + bias2; t = (t > 0.f) ? t : 0.01f * t;
        io[(size_t)(row0 + r + 8 * half) * D + col] = s + t;
    }
}

// ---------- host ----------
extern "C" void kernel_launch(void* const* d_in, const int* in_sizes, int n_in,
                              void* d_out, int out_size, void* d_ws, size_t ws_size,
                              hipStream_t stream) {
    const float* ego   = (const float*)d_in[0];
    const float* evals = (const float*)d_in[1];
    const float* W1    = (const float*)d_in[2];
    const float* b1    = (const float*)d_in[3];
    const float* W2    = (const float*)d_in[4];
    const float* b2    = (const float*)d_in[5];
    const int*   erows = (const int*)d_in[6];
    const int*   ecols = (const int*)d_in[7];
    float* out = (float*)d_out;
    unsigned short* wb = (unsigned short*)d_ws;   // 256 KB bf16 weights

    // 1) zero side accumulator (d_out doubles as side buffer)
    {
        int n4 = N_NODES * D / 4;                 // 6,400,000
        zero_f4<<<(n4 + 255) / 256, 256, 0, stream>>>((float4*)out, n4);
    }
    // 2) convert weights to bf16
    wconv_bf16<<<64, 256, 0, stream>>>(W1, W2, wb);
    // 3) SpMM: one wave per edge, 8 f32 atomics per lane
    {
        long long threads = (long long)N_EDGES * 32;
        int blocks = (int)((threads + 255) / 256);  // 400,000
        spmm_atomic<<<blocks, 256, 0, stream>>>(ego, evals, erows, ecols, out);
    }
    // 4) fused bi-interaction GEMMs with bf16 WMMA
    fused_bi_wmma<<<N_NODES / 16, 512, 0, stream>>>(ego, out, wb, b1, b2);
}
